// GET_Model_223338299536
// MI455X (gfx1250) — compile-verified, matrix operand order checked by reading, not run on hardware
//
#include <hip/hip_runtime.h>
#include <hip/hip_bf16.h>
#include <math.h>

typedef __attribute__((ext_vector_type(16))) _Float16 v16h;
typedef __attribute__((ext_vector_type(8)))  _Float16 v8h;
typedef __attribute__((ext_vector_type(8)))  float    v8f;
typedef long long ll;

#define HID 256
#define NCLS 5   // HW == HE == 5

// ============================================================================
// Generic dual-source batched WMMA GEMM:
//   C[b] = act( A1[b] @ W1[b]  +  (A2 ? A2[b] @ W2[b] : 0)  + bias1 + bias2 )
// A: [M,K] row-major (lda), W: [K,N] row-major (ldw), C: [M,N] (ldc).
// 128 threads = 4 waves; 64x64 C tile. A tile kept row-major-in-K in LDS,
// W tile stored TRANSPOSED ([n][k]) so both A and B fragments are loaded as
// two contiguous 128-bit ds_load_b128 per lane (matches the 16-bit A layout:
// lane(half) holds K runs {8h..8h+7} and {16+8h..23+8h}).
// Requires: K % 4 == 0, A base/lda 16-byte aligned (true for this model).
// ============================================================================
__global__ __launch_bounds__(128)
void wmma_gemm_kernel(const float* __restrict__ A1, ll sA1, int lda1,
                      const float* __restrict__ W1, ll sW1, int ldw1,
                      const float* __restrict__ A2, ll sA2, int lda2,
                      const float* __restrict__ W2, ll sW2, int ldw2,
                      const float* __restrict__ bias1, const float* __restrict__ bias2,
                      float* __restrict__ C, ll sC, int ldc,
                      int M, int N, int K, int act)
{
  __shared__ _Float16 As[64][40];    // [m][k], 80B row stride (16B aligned)
  __shared__ _Float16 WsT[64][40];   // [n][k], transposed W tile
  const int bz   = blockIdx.z;
  const int m0   = blockIdx.x * 64;
  const int n0   = blockIdx.y * 64;
  const int tid  = threadIdx.x;
  const int lane = tid & 31;
  const int wv   = tid >> 5;         // wave 0..3 -> rows [16*wv, 16*wv+16)
  const int lrow = lane & 15;
  const int lhalf = lane >> 4;

  v8f acc[4];
#pragma unroll
  for (int i = 0; i < 4; ++i)
#pragma unroll
    for (int j = 0; j < 8; ++j) acc[i][j] = 0.0f;

  const int npass = (A2 != nullptr) ? 2 : 1;
  for (int pass = 0; pass < npass; ++pass) {
    const float* Ap = pass ? (A2 + (ll)bz * sA2) : (A1 + (ll)bz * sA1);
    const float* Wp = pass ? (W2 + (ll)bz * sW2) : (W1 + (ll)bz * sW1);
    const int lda = pass ? lda2 : lda1;
    const int ldw = pass ? ldw2 : ldw1;
    for (int k0 = 0; k0 < K; k0 += 32) {
      // ---- stage A tile: thread owns a 16-float row segment ----
      {
        int r  = tid >> 1;            // 0..63
        int kb = (tid & 1) * 16;      // 0 or 16
        float vals[16];
        bool rowok = (m0 + r < M);
        const float* gp = Ap + (ll)(m0 + r) * lda + (k0 + kb);
#pragma unroll
        for (int i = 0; i < 4; ++i) {
          float4 f = make_float4(0.f, 0.f, 0.f, 0.f);
          if (rowok && (k0 + kb + i * 4) < K)
            f = *(const float4*)(gp + i * 4);
          vals[i * 4 + 0] = f.x; vals[i * 4 + 1] = f.y;
          vals[i * 4 + 2] = f.z; vals[i * 4 + 3] = f.w;
        }
        v8h h0, h1;
#pragma unroll
        for (int i = 0; i < 8; ++i) {
          h0[i] = (_Float16)vals[i];
          h1[i] = (_Float16)vals[8 + i];
        }
        *(v8h*)&As[r][kb]     = h0;
        *(v8h*)&As[r][kb + 8] = h1;
      }
      // ---- stage W tile transposed: coalesced global read, b16 LDS store ----
#pragma unroll
      for (int i = 0; i < 16; ++i) {
        int idx = tid + i * 128;
        int r = idx >> 6, c = idx & 63;          // r = k-in-tile, c = n-in-tile
        float v = 0.0f;
        if (k0 + r < K && n0 + c < N) v = Wp[(ll)(k0 + r) * ldw + (n0 + c)];
        WsT[c][r] = (_Float16)v;
      }
      __syncthreads();
      if (k0 + 32 < K)  // speculative prefetch of next A tile -> global_prefetch
        __builtin_prefetch(&Ap[(ll)(m0 + (tid >> 1)) * lda + (k0 + 32)], 0, 0);

      // ---- A fragment: two ds_load_b128 ----
      union { v16h v; v8h h[2]; } ua;
      {
        const v8h* ap = (const v8h*)&As[(wv << 4) + lrow][0];
        ua.h[0] = ap[lhalf];        // K = 8*half .. 8*half+7
        ua.h[1] = ap[lhalf + 2];    // K = 16+8*half .. 23+8*half
      }
#pragma unroll
      for (int nt = 0; nt < 4; ++nt) {
        union { v16h v; v8h h[2]; } ub;
        const v8h* bp = (const v8h*)&WsT[(nt << 4) + lrow][0];
        ub.h[0] = bp[lhalf];
        ub.h[1] = bp[lhalf + 2];
        acc[nt] = __builtin_amdgcn_wmma_f32_16x16x32_f16(
            false, ua.v, false, ub.v, (short)0, acc[nt], false, false);
      }
      __syncthreads();
    }
  }

  // epilogue: bias + activation + guarded store (C layout: vgpr r -> M, lane -> N)
  float* Cp = C + (ll)bz * sC;
#pragma unroll
  for (int nt = 0; nt < 4; ++nt) {
    int col = n0 + (nt << 4) + lrow;
    float bb = 0.0f;
    if (col < N) {
      if (bias1) bb += bias1[col];
      if (bias2) bb += bias2[col];
    }
#pragma unroll
    for (int r = 0; r < 8; ++r) {
      int row = m0 + (wv << 4) + (lhalf << 3) + r;
      if (row < M && col < N) {
        float v = acc[nt][r] + bb;
        if      (act == 1) v = 1.0f / (1.0f + expf(-v));   // sigmoid
        else if (act == 2) v = tanhf(v);
        else if (act == 3) v = fmaxf(v, 0.0f);             // relu
        Cp[(ll)row * ldc + col] = v;
      }
    }
  }
}

// ============================================================================
// Elementwise / small kernels
// ============================================================================
__global__ void gather_rows_kernel(const int* __restrict__ tok,
                                   const float* __restrict__ emb,
                                   float* __restrict__ out, ll n_rows) {
  ll gid = (ll)blockIdx.x * blockDim.x + threadIdx.x;
  if (gid >= n_rows * HID) return;
  ll row = gid / HID; int d = (int)(gid % HID);
  out[gid] = emb[(ll)tok[row] * HID + d];
}

__global__ void ew_mul_kernel(const float* __restrict__ a, const float* __restrict__ b,
                              float* __restrict__ out, ll n) {
  ll gid = (ll)blockIdx.x * blockDim.x + threadIdx.x;
  if (gid < n) out[gid] = a[gid] * b[gid];
}

// x <- h*z + x*(1-z)   (in-place over x)
__global__ void ggnn_combine_kernel(const float* __restrict__ h, const float* __restrict__ z,
                                    float* __restrict__ x, ll n) {
  ll gid = (ll)blockIdx.x * blockDim.x + threadIdx.x;
  if (gid < n) {
    float zz = z[gid];
    x[gid] = h[gid] * zz + x[gid] * (1.0f - zz);
  }
}

// out[row] = dot(A[row,:K], w[:K])     (g_score projection, dout=1)
__global__ void rowdot_kernel(const float* __restrict__ A, const float* __restrict__ w,
                              float* __restrict__ out, ll rows, int K) {
  ll gid = (ll)blockIdx.x * blockDim.x + threadIdx.x;
  if (gid >= rows) return;
  const float* a = A + gid * K;
  float s = 0.0f;
  for (int k = 0; k < K; ++k) s += a[k] * w[k];
  out[gid] = s;
}

// out[t,i] = sum_j adj[t,i,j] * x[t*L + j]
__global__ void batched_matvec_kernel(const float* __restrict__ adj, const float* __restrict__ x,
                                      float* __restrict__ out, int T, int L) {
  ll gid = (ll)blockIdx.x * blockDim.x + threadIdx.x;
  if (gid >= (ll)T * L) return;
  int t = (int)(gid / L), i = (int)(gid % L);
  const float* ar = adj + ((ll)t * L + i) * L;
  const float* xv = x + (ll)t * L;
  float s = 0.0f;
  for (int j = 0; j < L; ++j) s += ar[j] * xv[j];
  out[gid] = s;
}

// full scalar GGNN gate math for dout==1 (g_score)
__global__ void score_fuse_kernel(const float* __restrict__ xs, const float* __restrict__ as_,
    const float* z0W, const float* z0b, const float* z1W, const float* z1b,
    const float* r0W, const float* r0b, const float* r1W, const float* r1b,
    const float* h0W, const float* h0b, const float* h1W, const float* h1b,
    float* __restrict__ out, ll n) {
  ll gid = (ll)blockIdx.x * blockDim.x + threadIdx.x;
  if (gid >= n) return;
  float x = xs[gid], a = as_[gid];
  float z = 1.0f / (1.0f + expf(-(a * z0W[0] + z0b[0] + x * z1W[0] + z1b[0])));
  float r = 1.0f / (1.0f + expf(-(a * r0W[0] + r0b[0] + x * r1W[0] + r1b[0])));
  float h = tanhf(a * h0W[0] + h0b[0] + (r * x) * h1W[0] + h1b[0]);
  out[gid] = h * z + x * (1.0f - z);
}

// GSL: keep nodes whose rank < k (rank = #greater + #equal-with-smaller-index)
__global__ void topk_rowmask_kernel(const float* __restrict__ score, float* __restrict__ row,
                                    int L, int k) {
  __shared__ float s[256];
  int t = blockIdx.x, tid = threadIdx.x;
  if (tid < L) s[tid] = score[(ll)t * L + tid];
  __syncthreads();
  if (tid < L) {
    float si = s[tid];
    int rank = 0;
    for (int j = 0; j < L; ++j) {
      float sj = s[j];
      rank += (sj > si) || (sj == si && j < tid);
    }
    row[(ll)t * L + tid] = (rank < k) ? 1.0f : 0.0f;
  }
}

__global__ void mask_adj_kernel(const float* __restrict__ adj, const float* __restrict__ row,
                                float* __restrict__ out, int T, int L) {
  ll gid = (ll)blockIdx.x * blockDim.x + threadIdx.x;
  ll n = (ll)T * L * L;
  if (gid >= n) return;
  int t = (int)(gid / ((ll)L * L));
  int rem = (int)(gid % ((ll)L * L));
  int i = rem / L, j = rem % L;
  out[gid] = adj[gid] * fmaxf(row[(ll)t * L + i], row[(ll)t * L + j]);
}

// q_repr[b,d] = sum_l x[b,l,d]*mask[b,l] / sum_l mask[b,l]
__global__ void masked_mean_kernel(const float* __restrict__ x, const float* __restrict__ mask,
                                   float* __restrict__ out, int B, int L, int D) {
  ll gid = (ll)blockIdx.x * blockDim.x + threadIdx.x;
  if (gid >= (ll)B * D) return;
  int b = (int)(gid / D), d = (int)(gid % D);
  float num = 0.0f, den = 0.0f;
  for (int l = 0; l < L; ++l) {
    float m = mask[(ll)b * L + l];
    num += x[((ll)b * L + l) * D + d] * m;
    den += m;
  }
  out[gid] = num / den;
}

__global__ void starts_kernel(const int* __restrict__ counts, int B, int* __restrict__ starts) {
  if (threadIdx.x == 0 && blockIdx.x == 0) {
    int run = 0;
    for (int b = 0; b < B; ++b) { starts[b] = run; run += counts[b]; }
  }
}

__global__ void map_kernel(const int* __restrict__ counts, const int* __restrict__ starts,
                           int B, int T, int* __restrict__ d2q, int* __restrict__ slot) {
  int t = blockIdx.x * blockDim.x + threadIdx.x;
  if (t >= T) return;
  int bsel = B - 1;
  for (int b = 0; b < B; ++b)
    if (t >= starts[b] && t < starts[b] + counts[b]) { bsel = b; break; }
  d2q[t] = bsel;
  slot[t] = t - starts[bsel];
}

__global__ void evd_mask_kernel(const int* __restrict__ counts, float* __restrict__ em,
                                int B, int ME) {
  int gid = blockIdx.x * blockDim.x + threadIdx.x;
  if (gid >= B * ME) return;
  em[gid] = ((gid % ME) < counts[gid / ME]) ? 1.0f : 0.0f;
}

// lin[row,:] = tanh(G[row]+Q[q]) @ W2 ; masked rows forced to -1e30
__global__ void lin_kernel(const float* __restrict__ G, const float* __restrict__ Q,
                           const int* __restrict__ map, int Lrow,
                           const float* __restrict__ W2, const float* __restrict__ maskArr,
                           float* __restrict__ out, ll rows) {
  ll row = (ll)blockIdx.x * blockDim.x + threadIdx.x;
  if (row >= rows) return;
  int grp = (int)(row / Lrow);
  int q = map ? map[grp] : grp;
  float acc[NCLS];
#pragma unroll
  for (int h = 0; h < NCLS; ++h) acc[h] = 0.0f;
  for (int j = 0; j < 32; ++j) {
    float v = tanhf(G[row * 32 + j] + Q[(ll)q * 32 + j]);
#pragma unroll
    for (int h = 0; h < NCLS; ++h) acc[h] += v * W2[j * NCLS + h];
  }
  float m = maskArr[row];
#pragma unroll
  for (int h = 0; h < NCLS; ++h) out[row * NCLS + h] = (m == 0.0f) ? -1e30f : acc[h];
}

// softmax over L for each of NCLS heads; one block per group
__global__ void softmax_kernel(const float* __restrict__ lin, float* __restrict__ w, int L) {
  __shared__ float red[256];
  int t = blockIdx.x, tid = threadIdx.x;
  for (int h = 0; h < NCLS; ++h) {
    float v = (tid < L) ? lin[((ll)t * L + tid) * NCLS + h] : -1e30f;
    red[tid] = v; __syncthreads();
    for (int s = 128; s > 0; s >>= 1) { if (tid < s) red[tid] = fmaxf(red[tid], red[tid + s]); __syncthreads(); }
    float mx = red[0]; __syncthreads();
    float e = (tid < L) ? expf(v - mx) : 0.0f;
    red[tid] = e; __syncthreads();
    for (int s = 128; s > 0; s >>= 1) { if (tid < s) red[tid] += red[tid + s]; __syncthreads(); }
    float sm = red[0]; __syncthreads();
    if (tid < L) w[((ll)t * L + tid) * NCLS + h] = e / sm;
  }
}

// out[g, d*NCLS+h] = sum_l right[g,l,d] * w[g,l,h]
__global__ void attend_kernel(const float* __restrict__ right, const float* __restrict__ w,
                              float* __restrict__ out, int G, int L, int D) {
  ll gid = (ll)blockIdx.x * blockDim.x + threadIdx.x;
  if (gid >= (ll)G * D) return;
  int g = (int)(gid / D), d = (int)(gid % D);
  float acc[NCLS];
#pragma unroll
  for (int h = 0; h < NCLS; ++h) acc[h] = 0.0f;
  for (int l = 0; l < L; ++l) {
    float rv = right[((ll)g * L + l) * D + d];
    const float* wp = w + ((ll)g * L + l) * NCLS;
#pragma unroll
    for (int h = 0; h < NCLS; ++h) acc[h] += rv * wp[h];
  }
  float* op = out + ((ll)g * D + d) * NCLS;
#pragma unroll
  for (int h = 0; h < NCLS; ++h) op[h] = acc[h];
}

__global__ void zero_kernel(float* __restrict__ p, ll n) {
  ll gid = (ll)blockIdx.x * blockDim.x + threadIdx.x;
  if (gid < n) p[gid] = 0.0f;
}

__global__ void scatter_padded_kernel(const float* __restrict__ att, const int* __restrict__ d2q,
                                      const int* __restrict__ slot, float* __restrict__ padded,
                                      int T, int W, int ME) {
  ll gid = (ll)blockIdx.x * blockDim.x + threadIdx.x;
  if (gid >= (ll)T * W) return;
  int t = (int)(gid / W), j = (int)(gid % W);
  padded[((ll)d2q[t] * ME + slot[t]) * W + j] = att[gid];
}

__global__ void concat_qd_kernel(const float* __restrict__ q, const float* __restrict__ dr,
                                 float* __restrict__ qd, int B, int DW) {
  ll gid = (ll)blockIdx.x * blockDim.x + threadIdx.x;
  ll n = (ll)B * (HID + DW);
  if (gid >= n) return;
  int b = (int)(gid / (HID + DW)), j = (int)(gid % (HID + DW));
  qd[gid] = (j < HID) ? q[(ll)b * HID + j] : dr[(ll)b * DW + (j - HID)];
}

__global__ void final_out_kernel(const float* __restrict__ h, const float* __restrict__ W2,
                                 const float* __restrict__ b2, float* __restrict__ out, int B) {
  int gid = blockIdx.x * blockDim.x + threadIdx.x;
  if (gid >= B * 2) return;
  int b = gid / 2, c = gid % 2;
  float s = b2[c];
  for (int k = 0; k < HID; ++k) s += h[(ll)b * HID + k] * W2[k * 2 + c];
  out[gid] = s;
}

// ============================================================================
// Host-side helpers
// ============================================================================
struct GP { const float *h0W,*h0b,*h1W,*h1b,*proj,*r0W,*r0b,*r1W,*r1b,*z0W,*z0b,*z1W,*z1b; };
static GP gp(void* const* d_in, int base) {
  GP p;
  const float** f = (const float**)&p;
  for (int i = 0; i < 13; ++i) f[i] = (const float*)d_in[base + i];
  return p;
}

static inline dim3 g1(ll n) { return dim3((unsigned)((n + 255) / 256)); }

static void gemm(hipStream_t st,
                 const float* A1, ll sA1, int lda1, const float* W1, ll sW1, int ldw1,
                 const float* A2, ll sA2, int lda2, const float* W2, ll sW2, int ldw2,
                 const float* b1, const float* b2,
                 float* C, ll sC, int ldc, int M, int N, int K, int act, int batch) {
  dim3 grid((M + 63) / 64, (N + 63) / 64, batch);
  wmma_gemm_kernel<<<grid, dim3(128), 0, st>>>(A1, sA1, lda1, W1, sW1, ldw1,
                                               A2, sA2, lda2, W2, sW2, ldw2,
                                               b1, b2, C, sC, ldc, M, N, K, act);
}

// One inference GGNN (din -> 256). Output lands in-place in `x`.
static void run_ggnn(hipStream_t st, const float* adj, int batch, int L,
                     const float* xin, int din, const GP& p,
                     float* x, float* a, float* z, float* r, float* h) {
  int M = batch * L;
  ll n = (ll)M * HID;
  // x = xin @ proj
  gemm(st, xin, 0, din, p.proj, 0, HID, nullptr, 0, 0, nullptr, 0, 0,
       nullptr, nullptr, x, 0, HID, M, HID, din, 0, 1);
  // a = adj @ x (batched)
  gemm(st, adj, (ll)L * L, L, x, (ll)L * HID, HID, nullptr, 0, 0, nullptr, 0, 0,
       nullptr, nullptr, a, (ll)L * HID, HID, L, HID, L, 0, batch);
  // z = sigmoid(a@z0W + x@z1W + z0b + z1b)  — fused dual-source GEMM
  gemm(st, a, 0, HID, p.z0W, 0, HID, x, 0, HID, p.z1W, 0, HID,
       p.z0b, p.z1b, z, 0, HID, M, HID, HID, 1, 1);
  // r = sigmoid(a@r0W + x@r1W + ...)
  gemm(st, a, 0, HID, p.r0W, 0, HID, x, 0, HID, p.r1W, 0, HID,
       p.r0b, p.r1b, r, 0, HID, M, HID, HID, 1, 1);
  // r <- r*x (in place)
  ew_mul_kernel<<<g1(n), 256, 0, st>>>(r, x, r, n);
  // h = tanh(a@h0W + (r*x)@h1W + ...)
  gemm(st, a, 0, HID, p.h0W, 0, HID, r, 0, HID, p.h1W, 0, HID,
       p.h0b, p.h1b, h, 0, HID, M, HID, HID, 2, 1);
  // x <- h*z + x*(1-z)
  ggnn_combine_kernel<<<g1(n), 256, 0, st>>>(h, z, x, n);
}

extern "C" void kernel_launch(void* const* d_in, const int* in_sizes, int n_in,
                              void* d_out, int out_size, void* d_ws, size_t ws_size,
                              hipStream_t stream) {
  const int B = 32, Lq = 64, T = 960, L = 200, ME = 30, KKEEP = 100;
  const int DW = 1280;          // HID*HW (word-att width)
  const int DR = DW * NCLS;     // 6400
  const int QD = HID + DR;      // 6656

  const int*   query      = (const int*)  d_in[0];
  const float* query_adj  = (const float*)d_in[1];
  const float* query_mask = (const float*)d_in[2];
  const int*   doc        = (const int*)  d_in[3];
  const float* doc_adj    = (const float*)d_in[4];
  const float* doc_mask   = (const float*)d_in[5];
  const int*   counts     = (const int*)  d_in[6];
  const float* attE_W1    = (const float*)d_in[7];   // [1536,32]
  const float* attE_W2    = (const float*)d_in[8];   // [32,5]
  const float* attW_W1    = (const float*)d_in[9];   // [512,32]
  const float* attW_W2    = (const float*)d_in[10];  // [32,5]
  const float* emb        = (const float*)d_in[11];  // [1000,256]
  GP g_claim = gp(d_in, 12);
  GP g_fp1   = gp(d_in, 25);
  GP g_fp2   = gp(d_in, 38);
  GP g_sc    = gp(d_in, 51);
  const float* mlpW1 = (const float*)d_in[64];  // [6656,256]
  const float* mlpW2 = (const float*)d_in[65];  // [256,2]
  const float* mlpb1 = (const float*)d_in[66];
  const float* mlpb2 = (const float*)d_in[67];

  // ---- workspace carve ----
  char* base = (char*)d_ws; size_t off = 0;
  auto alloc = [&](ll elems) -> float* {
    float* p = (float*)(base + off);
    off += ((size_t)elems * 4 + 255) & ~(size_t)255;
    return p;
  };
  const ll U = (ll)T * L * HID;  // 49,152,000 floats
  float* Ue   = alloc(U);              // emb_d, later h-buffer
  float* Ux1  = alloc(U);              // ggnn1 x -> f1
  float* Ux2  = alloc(U);              // ggnn2 x -> doc node features
  float* Ta   = alloc(U);
  float* Tz   = alloc(U);
  float* Tr   = alloc(U);
  float* adjR = alloc((ll)T * L * L);
  float* xs    = alloc((ll)T * L);
  float* as_   = alloc((ll)T * L);
  float* score = alloc((ll)T * L);
  float* rowm  = alloc((ll)T * L);
  float* Gw    = alloc((ll)T * L * 32);
  float* lin   = alloc((ll)T * L * NCLS);
  float* wsm   = alloc((ll)T * L * NCLS);
  float* attd  = alloc((ll)T * DW);
  float* padded= alloc((ll)B * ME * DW);
  float* Q1    = alloc(B * 32);
  float* Q2    = alloc(B * 32);
  float* G2    = alloc((ll)B * ME * 32);
  float* lin2  = alloc((ll)B * ME * NCLS);
  float* w2    = alloc((ll)B * ME * NCLS);
  float* docr  = alloc((ll)B * DR);
  float* qd    = alloc((ll)B * QD);
  float* mlph  = alloc((ll)B * HID);
  float* qemb  = alloc((ll)B * Lq * HID);
  float* qx    = alloc((ll)B * Lq * HID);
  float* qa    = alloc((ll)B * Lq * HID);
  float* qz    = alloc((ll)B * Lq * HID);
  float* qr    = alloc((ll)B * Lq * HID);
  float* qh    = alloc((ll)B * Lq * HID);
  float* qrep  = alloc((ll)B * HID);
  float* evm   = alloc(B * ME);
  int* starts = (int*)alloc(B);
  int* d2q    = (int*)alloc(T);
  int* slot   = (int*)alloc(T);

  // ---- query branch ----
  gather_rows_kernel<<<g1((ll)B * Lq * HID), 256, 0, stream>>>(query, emb, qemb, (ll)B * Lq);
  run_ggnn(stream, query_adj, B, Lq, qemb, HID, g_claim, qx, qa, qz, qr, qh);
  masked_mean_kernel<<<g1((ll)B * HID), 256, 0, stream>>>(qx, query_mask, qrep, B, Lq, HID);

  // ---- evidence mapping ----
  starts_kernel<<<1, 1, 0, stream>>>(counts, B, starts);
  map_kernel<<<g1(T), 256, 0, stream>>>(counts, starts, B, T, d2q, slot);
  evd_mask_kernel<<<g1(B * ME), 256, 0, stream>>>(counts, evm, B, ME);

  // ---- doc branch: GGNN fp1 ----
  gather_rows_kernel<<<g1(U), 256, 0, stream>>>(doc, emb, Ue, (ll)T * L);
  run_ggnn(stream, doc_adj, T, L, Ue, HID, g_fp1, Ux1, Ta, Tz, Tr, Ue); // f1 -> Ux1

  // ---- g_score GGNN (dout = 1) ----
  rowdot_kernel<<<g1((ll)T * L), 256, 0, stream>>>(Ux1, g_sc.proj, xs, (ll)T * L, HID);
  batched_matvec_kernel<<<g1((ll)T * L), 256, 0, stream>>>(doc_adj, xs, as_, T, L);
  score_fuse_kernel<<<g1((ll)T * L), 256, 0, stream>>>(
      xs, as_, g_sc.z0W, g_sc.z0b, g_sc.z1W, g_sc.z1b,
      g_sc.r0W, g_sc.r0b, g_sc.r1W, g_sc.r1b,
      g_sc.h0W, g_sc.h0b, g_sc.h1W, g_sc.h1b, score, (ll)T * L);

  // ---- GSL ----
  topk_rowmask_kernel<<<T, 256, 0, stream>>>(score, rowm, L, KKEEP);
  mask_adj_kernel<<<g1((ll)T * L * L), 256, 0, stream>>>(doc_adj, rowm, adjR, T, L);

  // ---- GGNN fp2 on sparsified graph ----
  run_ggnn(stream, adjR, T, L, Ux1, HID, g_fp2, Ux2, Ta, Tz, Tr, Ue); // d_out -> Ux2

  // ---- word-level attention ----
  gemm(stream, qrep, 0, HID, attW_W1, 0, 32, nullptr, 0, 0, nullptr, 0, 0,
       nullptr, nullptr, Q1, 0, 32, B, 32, HID, 0, 1);
  gemm(stream, Ux2, 0, HID, attW_W1 + (ll)HID * 32, 0, 32, nullptr, 0, 0, nullptr, 0, 0,
       nullptr, nullptr, Gw, 0, 32, T * L, 32, HID, 0, 1);
  lin_kernel<<<g1((ll)T * L), 256, 0, stream>>>(Gw, Q1, d2q, L, attW_W2, doc_mask, lin, (ll)T * L);
  softmax_kernel<<<T, 256, 0, stream>>>(lin, wsm, L);
  attend_kernel<<<g1((ll)T * HID), 256, 0, stream>>>(Ux2, wsm, attd, T, L, HID);

  // ---- pad per-query evidence ----
  zero_kernel<<<g1((ll)B * ME * DW), 256, 0, stream>>>(padded, (ll)B * ME * DW);
  scatter_padded_kernel<<<g1((ll)T * DW), 256, 0, stream>>>(attd, d2q, slot, padded, T, DW, ME);

  // ---- evidence-level attention ----
  gemm(stream, qrep, 0, HID, attE_W1, 0, 32, nullptr, 0, 0, nullptr, 0, 0,
       nullptr, nullptr, Q2, 0, 32, B, 32, HID, 0, 1);
  gemm(stream, padded, 0, DW, attE_W1 + (ll)HID * 32, 0, 32, nullptr, 0, 0, nullptr, 0, 0,
       nullptr, nullptr, G2, 0, 32, B * ME, 32, DW, 0, 1);
  lin_kernel<<<g1((ll)B * ME), 256, 0, stream>>>(G2, Q2, nullptr, ME, attE_W2, evm, lin2, (ll)B * ME);
  softmax_kernel<<<B, 256, 0, stream>>>(lin2, w2, ME);
  attend_kernel<<<g1((ll)B * DW), 256, 0, stream>>>(padded, w2, docr, B, ME, DW);

  // ---- classifier head ----
  concat_qd_kernel<<<g1((ll)B * QD), 256, 0, stream>>>(qrep, docr, qd, B, DR);
  gemm(stream, qd, 0, QD, mlpW1, 0, HID, nullptr, 0, 0, nullptr, 0, 0,
       mlpb1, nullptr, mlph, 0, HID, B, HID, QD, 3, 1);
  final_out_kernel<<<g1(B * 2), 256, 0, stream>>>(mlph, mlpW2, mlpb2, (float*)d_out, B);

  (void)in_sizes; (void)n_in; (void)out_size; (void)ws_size;
}
